// ScaledDotProductAttention_41120016892212
// MI455X (gfx1250) — compile-verified, matrix operand order checked by reading, not run on hardware
//
#include <hip/hip_runtime.h>

// ---- CDNA5 WMMA types ----
typedef __attribute__((ext_vector_type(16))) __bf16 v16bf;
typedef __attribute__((ext_vector_type(4)))  __bf16 v4bf;
typedef __attribute__((ext_vector_type(8)))  float  v8f;
typedef int v4i_vs __attribute__((vector_size(16)));   // matches async builtin's pointee type

#define S_LEN 2048
#define D_DIM 128
#define BH    32        // B*H = 2*16
#define BM    128       // query rows per workgroup (8 waves x 16)
#define BN    128       // key cols per iteration
#define NWAVE 8
#define LK    144       // padded bf16 row pitch (288B = 9*32B: aligned + bank spread)

// softmax in base-2 domain: fold 1/T and log2(e) into the logit scale
#define SCALE2 1.8033688011112042f        // 1.25 * log2(e)
#define FILL2  (-1.4426950408889634e-9f)  // -1e-9 * log2(e)  (faithful NEG_FILL, log2 domain)

// ---- main-kernel LDS layout (bytes), double-buffered K / V^T tiles ----
#define TILE_B  (BN * LK * 2)                 // 36864
#define OFF_K0  0
#define OFF_K1  (OFF_K0 + TILE_B)
#define OFF_V0  (OFF_K1 + TILE_B)
#define OFF_V1  (OFF_V0 + TILE_B)
#define OFF_P   (OFF_V1 + TILE_B)
#define SMEM_MAIN (OFF_P + NWAVE * 16 * LK * 2)   // 184320

// ---- workspace layout: bf16 K (row-major) + bf16 V^T ([bh][d][s]) ----
#define WS_KB   0ull
#define WS_VT   ((unsigned long long)BH * S_LEN * D_DIM * 2ull)   // 16777216
#define WS_NEED (2ull * WS_VT)                                    // 33554432

__device__ __forceinline__ float fast_exp2(float x) {
#if __has_builtin(__builtin_amdgcn_exp2f)
    return __builtin_amdgcn_exp2f(x);       // native v_exp_f32
#else
    return exp2f(x);
#endif
}

// =============== gfx1250 async global->LDS copy (ASYNCcnt) ===============
#if __has_builtin(__builtin_amdgcn_global_load_async_to_lds_b128)
#define ASYNC_LDS 1
__device__ __forceinline__ void cp_async_b128(unsigned lds_off, const void* g) {
    __builtin_amdgcn_global_load_async_to_lds_b128(
        (__attribute__((address_space(1))) v4i_vs*)(unsigned long long)g,
        (__attribute__((address_space(3))) v4i_vs*)lds_off,
        0, 0);
}
__device__ __forceinline__ void wait_async0() {
#if __has_builtin(__builtin_amdgcn_s_wait_asynccnt)
    __builtin_amdgcn_s_wait_asynccnt(0);
#else
    asm volatile("s_wait_asynccnt 0x0" ::: "memory");
#endif
}
#else
#define ASYNC_LDS 0
__device__ __forceinline__ void wait_async0() {}
#endif

// ======================= pre-pass kernels =======================
__global__ __launch_bounds__(256)
void convert_k_bf16(const float* __restrict__ src, __bf16* __restrict__ dst) {
    size_t i = ((size_t)blockIdx.x * 256 + threadIdx.x) * 4;
    float4 f = *(const float4*)(src + i);
    v4bf b = { (__bf16)f.x, (__bf16)f.y, (__bf16)f.z, (__bf16)f.w };
    *(v4bf*)(dst + i) = b;
}

__global__ __launch_bounds__(256)
void transpose_v_bf16(const float* __restrict__ V, __bf16* __restrict__ Vt) {
    __shared__ float tile[32][33];
    const int bh = blockIdx.z;
    const int s0 = blockIdx.x * 32;
    const int d0 = blockIdx.y * 32;
    const int tx = threadIdx.x;    // 0..31
    const int ty = threadIdx.y;    // 0..7
    const float* vp = V + (size_t)bh * S_LEN * D_DIM;
#pragma unroll
    for (int i = ty; i < 32; i += 8)
        tile[i][tx] = vp[(size_t)(s0 + i) * D_DIM + d0 + tx];
    __syncthreads();
    __bf16* op = Vt + (size_t)bh * D_DIM * S_LEN;
#pragma unroll
    for (int i = ty; i < 32; i += 8)
        op[(size_t)(d0 + i) * S_LEN + s0 + tx] = (__bf16)tile[tx][i];
}

// ======================= main flash-attention kernel =======================
__global__ __launch_bounds__(256)
void fa_main_bf16_wmma(const __bf16* __restrict__ Kb,   // [bh][s][d] bf16
                       const __bf16* __restrict__ Vtb,  // [bh][d][s] bf16
                       const float*  __restrict__ Qg,   // [bh][s][d] f32
                       const int*    __restrict__ Mg,   // [s][s]
                       float*        __restrict__ Og)
{
    extern __shared__ char smem[];
    const unsigned ldsBase = (unsigned)(unsigned long long)(void*)smem; // aperture low 32b = LDS offset

    const int tid  = threadIdx.x;
    const int wv   = tid >> 5;
    const int lane = tid & 31;
    const int ln16 = lane & 15;
    const int hf   = lane >> 4;

    const int bh   = blockIdx.y;
    const int qBlk = blockIdx.x * BM;
    const size_t base = (size_t)bh * S_LEN * D_DIM;

    // ---- Q fragments (A layout: row = lane%16, 16 contig K, half-split) ----
    v16bf qf[4];
    {
        const int qRow = qBlk + wv * 16 + ln16;
        const float* qp = Qg + base + (size_t)qRow * D_DIM;
#pragma unroll
        for (int kc = 0; kc < 4; ++kc) {
            const float4* p4 = (const float4*)(qp + kc * 32 + hf * 16);
            v16bf a;
#pragma unroll
            for (int i = 0; i < 4; ++i) {
                float4 f = p4[i];
                a[i * 4 + 0] = (__bf16)f.x; a[i * 4 + 1] = (__bf16)f.y;
                a[i * 4 + 2] = (__bf16)f.z; a[i * 4 + 3] = (__bf16)f.w;
            }
            qf[kc] = a;
        }
    }

    // stage tile jn into buffer b: 128 rows x 256B, 16B chunks, 8 per thread each for K and V^T
    auto stage = [&](int jn, int b) {
        const unsigned kOff = ldsBase + (unsigned)(b ? OFF_K1 : OFF_K0);
        const unsigned vOff = ldsBase + (unsigned)(b ? OFF_V1 : OFF_V0);
        const char* kSrc = (const char*)(Kb + (base + (size_t)jn * BN * D_DIM));
        const char* vSrc = (const char*)(Vtb + (size_t)bh * D_DIM * S_LEN + (size_t)jn * BN);
#pragma unroll
        for (int i = 0; i < 8; ++i) {
            int chunk = i * 256 + tid;          // 0..2047
            int row   = chunk >> 4;             // 0..127
            int c16   = chunk & 15;             // 16B unit in row
            unsigned dOff = (unsigned)(row * (LK * 2) + c16 * 16);
#if ASYNC_LDS
            cp_async_b128(kOff + dOff, kSrc + row * (D_DIM * 2) + c16 * 16);
            cp_async_b128(vOff + dOff, vSrc + (size_t)row * (S_LEN * 2) + c16 * 16);
#else
            *(uint4*)(smem + (b ? OFF_K1 : OFF_K0) + dOff) =
                *(const uint4*)(kSrc + row * (D_DIM * 2) + c16 * 16);
            *(uint4*)(smem + (b ? OFF_V1 : OFF_V0) + dOff) =
                *(const uint4*)(vSrc + (size_t)row * (S_LEN * 2) + c16 * 16);
#endif
        }
    };

    // ---- accumulators & online-softmax state (row = hf*8 + r), base-2 logit domain ----
    v8f o[8];
    float m_i[8], l_i[8];
#pragma unroll
    for (int dt = 0; dt < 8; ++dt) o[dt] = v8f{};
#pragma unroll
    for (int r = 0; r < 8; ++r) { m_i[r] = -3.0e38f; l_i[r] = 0.0f; }

    __bf16* Pw = (__bf16*)(smem + OFF_P) + wv * (16 * LK);

    stage(0, 0);   // prologue

    for (int j = 0; j < S_LEN / BN; ++j) {
        const int cur = j & 1;
        wait_async0();           // this wave's tile-j copies arrived
        __syncthreads();         // all waves' copies arrived; prior-tile readers done
        if (j + 1 < S_LEN / BN) stage(j + 1, cur ^ 1);   // copies overlap compute below

        const __bf16* Ksm = (const __bf16*)(smem + (cur ? OFF_K1 : OFF_K0));
        const __bf16* Vts = (const __bf16*)(smem + (cur ? OFF_V1 : OFF_V0));

        // ================= S = (Q K^T) * (log2e/T), masked (mask straight from L2) =================
        v8f sf[8];
#pragma unroll
        for (int nt = 0; nt < 8; ++nt) {
            v8f acc = v8f{};
#pragma unroll
            for (int kc = 0; kc < 4; ++kc) {
                v16bf b = *(const v16bf*)(Ksm + (nt * 16 + ln16) * LK + kc * 32 + hf * 16);
                acc = __builtin_amdgcn_wmma_f32_16x16x32_bf16(
                          false, qf[kc], false, b, (short)0, acc, false, false);
            }
            sf[nt] = acc;
        }
        const int* Mrow = Mg + (size_t)(qBlk + wv * 16 + hf * 8) * S_LEN + j * BN;
#pragma unroll
        for (int nt = 0; nt < 8; ++nt) {
#pragma unroll
            for (int r = 0; r < 8; ++r) {
                float s = sf[nt][r] * SCALE2;
                if (Mrow[(size_t)r * S_LEN + nt * 16 + ln16] == 0)
                    s = FILL2;
                sf[nt][r] = s;
            }
        }

        // ================= online softmax (registers + shfl_xor width16, exp2 domain) =================
        float scl[8];
#pragma unroll
        for (int r = 0; r < 8; ++r) {
            float mx = sf[0][r];
#pragma unroll
            for (int nt = 1; nt < 8; ++nt) mx = fmaxf(mx, sf[nt][r]);
#pragma unroll
            for (int d = 1; d < 16; d <<= 1) mx = fmaxf(mx, __shfl_xor(mx, d, 16));
            float mnew = fmaxf(m_i[r], mx);
            scl[r] = fast_exp2(m_i[r] - mnew);
            m_i[r] = mnew;
        }
#pragma unroll
        for (int r = 0; r < 8; ++r) {
            float ls = 0.0f;
#pragma unroll
            for (int nt = 0; nt < 8; ++nt) {
                float p = fast_exp2(sf[nt][r] - m_i[r]);
                sf[nt][r] = p;
                ls += p;
            }
#pragma unroll
            for (int d = 1; d < 16; d <<= 1) ls += __shfl_xor(ls, d, 16);
            l_i[r] = l_i[r] * scl[r] + ls;
#pragma unroll
            for (int dt = 0; dt < 8; ++dt) o[dt][r] *= scl[r];
        }

        // ======== P: C-layout -> A-layout via per-wave LDS scratch (same-wave, in-order) ========
#pragma unroll
        for (int nt = 0; nt < 8; ++nt)
#pragma unroll
            for (int r = 0; r < 8; ++r)
                Pw[(hf * 8 + r) * LK + nt * 16 + ln16] = (__bf16)sf[nt][r];
        v16bf pf[4];
#pragma unroll
        for (int kc = 0; kc < 4; ++kc)
            pf[kc] = *(const v16bf*)(Pw + ln16 * LK + kc * 32 + hf * 16);

        // ================= O += P V =================
#pragma unroll
        for (int dt = 0; dt < 8; ++dt) {
            v8f acc = o[dt];
#pragma unroll
            for (int kc = 0; kc < 4; ++kc) {
                v16bf b = *(const v16bf*)(Vts + (dt * 16 + ln16) * LK + kc * 32 + hf * 16);
                acc = __builtin_amdgcn_wmma_f32_16x16x32_bf16(
                          false, pf[kc], false, b, (short)0, acc, false, false);
            }
            o[dt] = acc;
        }
    }

    // ================= epilogue: O / l =================
#pragma unroll
    for (int r = 0; r < 8; ++r) {
        float inv = 1.0f / l_i[r];
        int row = qBlk + wv * 16 + hf * 8 + r;
        float* op = Og + base + (size_t)row * D_DIM;
#pragma unroll
        for (int dt = 0; dt < 8; ++dt)
            op[dt * 16 + ln16] = o[dt][r] * inv;
    }
}

// ======================= fallback (used if ws too small) =======================
#define LM    132
#define FOFF_K   0
#define FOFF_VT  (FOFF_K  + BN * LK * 2)
#define FOFF_P   (FOFF_VT + D_DIM * LK * 2)
#define FOFF_MSK (FOFF_P  + NWAVE * 16 * LK * 2)
#define SMEM_FB  (FOFF_MSK + BM * LM)

__global__ __launch_bounds__(256)
void fa_fwd_fallback(const float* __restrict__ Kg, const float* __restrict__ Vg,
                     const float* __restrict__ Qg, const int* __restrict__ Mg,
                     float* __restrict__ Og)
{
    extern __shared__ char smem[];
    __bf16*        Ksm = (__bf16*)(smem + FOFF_K);
    __bf16*        Vts = (__bf16*)(smem + FOFF_VT);
    __bf16*        Psm = (__bf16*)(smem + FOFF_P);
    unsigned char* Msm = (unsigned char*)(smem + FOFF_MSK);

    const int tid  = threadIdx.x;
    const int wv   = tid >> 5;
    const int lane = tid & 31;
    const int ln16 = lane & 15;
    const int hf   = lane >> 4;
    const int bh   = blockIdx.y;
    const int qBlk = blockIdx.x * BM;
    const size_t base = (size_t)bh * S_LEN * D_DIM;

    v16bf qf[4];
    {
        const int qRow = qBlk + wv * 16 + ln16;
        const float* qp = Qg + base + (size_t)qRow * D_DIM;
#pragma unroll
        for (int kc = 0; kc < 4; ++kc) {
            const float4* p4 = (const float4*)(qp + kc * 32 + hf * 16);
            v16bf a;
#pragma unroll
            for (int i = 0; i < 4; ++i) {
                float4 f = p4[i];
                a[i * 4 + 0] = (__bf16)f.x; a[i * 4 + 1] = (__bf16)f.y;
                a[i * 4 + 2] = (__bf16)f.z; a[i * 4 + 3] = (__bf16)f.w;
            }
            qf[kc] = a;
        }
    }

    v8f o[8];
    float m_i[8], l_i[8];
#pragma unroll
    for (int dt = 0; dt < 8; ++dt) o[dt] = v8f{};
#pragma unroll
    for (int r = 0; r < 8; ++r) { m_i[r] = -3.0e38f; l_i[r] = 0.0f; }
    __bf16* Pw = Psm + wv * (16 * LK);

    for (int j = 0; j < S_LEN / BN; ++j) {
#pragma unroll 4
        for (int i = 0; i < 16; ++i) {
            int f = i * 256 + tid;
            int r = f >> 5;
            int c = (f & 31) * 4;
            {
                float4 kv = *(const float4*)(Kg + base + (size_t)(j * BN + r) * D_DIM + c);
                v4bf b4 = { (__bf16)kv.x, (__bf16)kv.y, (__bf16)kv.z, (__bf16)kv.w };
                *(v4bf*)(Ksm + r * LK + c) = b4;
            }
            {
                float4 vv = *(const float4*)(Vg + base + (size_t)(j * BN + r) * D_DIM + c);
                Vts[(c + 0) * LK + r] = (__bf16)vv.x;
                Vts[(c + 1) * LK + r] = (__bf16)vv.y;
                Vts[(c + 2) * LK + r] = (__bf16)vv.z;
                Vts[(c + 3) * LK + r] = (__bf16)vv.w;
            }
            {
                int4 mv = *(const int4*)(Mg + (size_t)(qBlk + r) * S_LEN + j * BN + c);
                Msm[r * LM + c + 0] = (unsigned char)(mv.x != 0);
                Msm[r * LM + c + 1] = (unsigned char)(mv.y != 0);
                Msm[r * LM + c + 2] = (unsigned char)(mv.z != 0);
                Msm[r * LM + c + 3] = (unsigned char)(mv.w != 0);
            }
        }
        __syncthreads();

        v8f sf[8];
#pragma unroll
        for (int nt = 0; nt < 8; ++nt) {
            v8f acc = v8f{};
#pragma unroll
            for (int kc = 0; kc < 4; ++kc) {
                v16bf b = *(const v16bf*)(Ksm + (nt * 16 + ln16) * LK + kc * 32 + hf * 16);
                acc = __builtin_amdgcn_wmma_f32_16x16x32_bf16(
                          false, qf[kc], false, b, (short)0, acc, false, false);
            }
            sf[nt] = acc;
        }
#pragma unroll
        for (int nt = 0; nt < 8; ++nt)
#pragma unroll
            for (int r = 0; r < 8; ++r) {
                float s = sf[nt][r] * SCALE2;
                int mrow = wv * 16 + hf * 8 + r;
                if (!Msm[mrow * LM + nt * 16 + ln16]) s = FILL2;
                sf[nt][r] = s;
            }

        float scl[8];
#pragma unroll
        for (int r = 0; r < 8; ++r) {
            float mx = sf[0][r];
#pragma unroll
            for (int nt = 1; nt < 8; ++nt) mx = fmaxf(mx, sf[nt][r]);
#pragma unroll
            for (int d = 1; d < 16; d <<= 1) mx = fmaxf(mx, __shfl_xor(mx, d, 16));
            float mnew = fmaxf(m_i[r], mx);
            scl[r] = fast_exp2(m_i[r] - mnew);
            m_i[r] = mnew;
        }
#pragma unroll
        for (int r = 0; r < 8; ++r) {
            float ls = 0.0f;
#pragma unroll
            for (int nt = 0; nt < 8; ++nt) {
                float p = fast_exp2(sf[nt][r] - m_i[r]);
                sf[nt][r] = p;
                ls += p;
            }
#pragma unroll
            for (int d = 1; d < 16; d <<= 1) ls += __shfl_xor(ls, d, 16);
            l_i[r] = l_i[r] * scl[r] + ls;
#pragma unroll
            for (int dt = 0; dt < 8; ++dt) o[dt][r] *= scl[r];
        }

#pragma unroll
        for (int nt = 0; nt < 8; ++nt)
#pragma unroll
            for (int r = 0; r < 8; ++r)
                Pw[(hf * 8 + r) * LK + nt * 16 + ln16] = (__bf16)sf[nt][r];
        v16bf pf[4];
#pragma unroll
        for (int kc = 0; kc < 4; ++kc)
            pf[kc] = *(const v16bf*)(Pw + ln16 * LK + kc * 32 + hf * 16);

#pragma unroll
        for (int dt = 0; dt < 8; ++dt) {
            v8f acc = o[dt];
#pragma unroll
            for (int kc = 0; kc < 4; ++kc) {
                v16bf b = *(const v16bf*)(Vts + (dt * 16 + ln16) * LK + kc * 32 + hf * 16);
                acc = __builtin_amdgcn_wmma_f32_16x16x32_bf16(
                          false, pf[kc], false, b, (short)0, acc, false, false);
            }
            o[dt] = acc;
        }
        __syncthreads();
    }

#pragma unroll
    for (int r = 0; r < 8; ++r) {
        float inv = 1.0f / l_i[r];
        int row = qBlk + wv * 16 + hf * 8 + r;
        float* op = Og + base + (size_t)row * D_DIM;
#pragma unroll
        for (int dt = 0; dt < 8; ++dt)
            op[dt * 16 + ln16] = o[dt][r] * inv;
    }
}

// ======================= launch =======================
extern "C" void kernel_launch(void* const* d_in, const int* in_sizes, int n_in,
                              void* d_out, int out_size, void* d_ws, size_t ws_size,
                              hipStream_t stream) {
    (void)in_sizes; (void)n_in; (void)out_size;
    const float* Kg = (const float*)d_in[0];
    const float* Vg = (const float*)d_in[1];
    const float* Qg = (const float*)d_in[2];
    const int*   Mg = (const int*)d_in[3];
    float*       Og = (float*)d_out;

    dim3 grid(S_LEN / BM, BH);
    dim3 block(256);

    if (ws_size >= (size_t)WS_NEED) {
        __bf16* Kb = (__bf16*)d_ws;
        __bf16* Vt = (__bf16*)((char*)d_ws + WS_VT);
        // pre-pass: K -> bf16 row-major; V -> bf16 transposed [bh][d][s]
        convert_k_bf16<<<(BH * S_LEN * D_DIM) / 4 / 256, 256, 0, stream>>>(Kg, Kb);
        transpose_v_bf16<<<dim3(S_LEN / 32, D_DIM / 32, BH), dim3(32, 8), 0, stream>>>(Vg, Vt);
        fa_main_bf16_wmma<<<grid, block, SMEM_MAIN, stream>>>(Kb, Vt, Qg, Mg, Og);
    } else {
        fa_fwd_fallback<<<grid, block, SMEM_FB, stream>>>(Kg, Vg, Qg, Mg, Og);
    }
}
